// AV_MiltiHeadAttention_59828894433541
// MI455X (gfx1250) — matrix operand order repaired and checked by reading, not compiled, not run on hardware
//
#include <hip/hip_runtime.h>
#include <cstdint>

// ---------------------------------------------------------------------------
// CDNA5 / gfx1250 bf16-WMMA implementation of AV_MultiHeadAttention forward.
// All GEMMs are NT-form with two contiguous 16B fragment loads per lane (symmetric-matrix
// and transposed-output tricks remove every strided gather). kT transpose
// stages through LDS via global_load_async_to_lds_b128 (ASYNCcnt path).
// ---------------------------------------------------------------------------

typedef __attribute__((ext_vector_type(16))) __bf16 v16bf;
typedef __attribute__((ext_vector_type(8)))  float  v8f;

#define B_  8
#define N_  1024
#define C_  512
#define H_  8
#define HD_ 64
#define M_  (B_ * N_)   // 8192 token rows

// ------------------------- scalar conversion helpers -----------------------
__device__ __forceinline__ unsigned short f2bf(float f) {
  union { float f; unsigned int u; } c; c.f = f;
  unsigned int u = c.u;
  u += 0x7FFFu + ((u >> 16) & 1u);      // round-to-nearest-even
  return (unsigned short)(u >> 16);
}

// ------------------------------ WMMA wrapper -------------------------------
__device__ __forceinline__ v8f wmma_bf16(v16bf a, v16bf b, v8f c) {
  return __builtin_amdgcn_wmma_f32_16x16x32_bf16(
      false, a, false, b, (short)0, c, false, false);
}

// --------------------------- fragment loaders ------------------------------
// A fragment (16x32 bf16) from row-major [rows, ld] at (row0, k0).
__device__ __forceinline__ v16bf load_frag_a(const unsigned short* src, int ld,
                                             int row0, int k0, int lane) {
  int r = lane & 15, h = lane >> 4;
  const unsigned short* p = src + (size_t)(row0 + r) * ld + k0;
  union { v16bf v; uint4 q[2]; } a;
  a.q[0] = *(const uint4*)(p + 8 * h);
  a.q[1] = *(const uint4*)(p + 16 + 8 * h);
  return a.v;
}

// B fragment for NT gemm (out = X @ W^T): B[k][n] = W[n][k], W row-major [Nt, ld].
__device__ __forceinline__ v16bf load_frag_b_nt(const unsigned short* w, int ld,
                                                int col0, int k0, int lane) {
  int n = lane & 15, h = lane >> 4;
  const unsigned short* p = w + (size_t)(col0 + n) * ld + k0;
  union { v16bf v; uint4 q[2]; } b;
  b.q[0] = *(const uint4*)(p + 8 * h);
  b.q[1] = *(const uint4*)(p + 16 + 8 * h);
  return b.v;
}

// ------------------------------ kernels ------------------------------------
__global__ __launch_bounds__(256) void k_f2bf(const float* __restrict__ in,
                                              unsigned short* __restrict__ out,
                                              int n) {
  int i = blockIdx.x * blockDim.x + threadIdx.x;
  if (i < n) out[i] = f2bf(in[i]);
}

// Y[M,Nt] = X[M,K] @ W[Nt,K]^T (+bias). WG tile 64x128, 8 waves, 4 wmma acc each.
template <bool OUT_BF16, bool BIAS>
__global__ __launch_bounds__(256) void k_gemm_nt(
    const unsigned short* __restrict__ X, const unsigned short* __restrict__ W,
    void* __restrict__ Y, const float* __restrict__ bias, int M, int Nt, int K) {
  int lane = threadIdx.x & 31, w = threadIdx.x >> 5;
  int m0 = blockIdx.x * 64 + (w >> 2) * 32;
  int n0 = blockIdx.y * 128 + (w & 3) * 32;
  v8f acc[2][2] = {};
  for (int k0 = 0; k0 < K; k0 += 32) {
    v16bf a0 = load_frag_a(X, K, m0, k0, lane);
    v16bf a1 = load_frag_a(X, K, m0 + 16, k0, lane);
    v16bf b0 = load_frag_b_nt(W, K, n0, k0, lane);
    v16bf b1 = load_frag_b_nt(W, K, n0 + 16, k0, lane);
    acc[0][0] = wmma_bf16(a0, b0, acc[0][0]);
    acc[0][1] = wmma_bf16(a0, b1, acc[0][1]);
    acc[1][0] = wmma_bf16(a1, b0, acc[1][0]);
    acc[1][1] = wmma_bf16(a1, b1, acc[1][1]);
  }
  int hh = lane >> 4, nn = lane & 15;
#pragma unroll
  for (int i = 0; i < 2; ++i)
#pragma unroll
    for (int j = 0; j < 2; ++j) {
      int row0 = m0 + 16 * i, col0 = n0 + 16 * j;
      float bv = BIAS ? bias[col0 + nn] : 0.0f;
#pragma unroll
      for (int t = 0; t < 8; ++t) {
        float v = acc[i][j][t] + bv;
        size_t off = (size_t)(row0 + 8 * hh + t) * Nt + col0 + nn;
        if (OUT_BF16) ((unsigned short*)Y)[off] = f2bf(v);
        else          ((float*)Y)[off] = v;
      }
    }
}

// qs GEMM with transposed bf16 output: Yt[b][col][n] (ld = N_), so that
// qs^T @ qs becomes a contiguous NT gemm. Transposed store is contiguous/lane.
__global__ __launch_bounds__(256) void k_gemm_nt_tout(
    const unsigned short* __restrict__ X, const unsigned short* __restrict__ W,
    unsigned short* __restrict__ Yt, int Nt, int K) {
  int lane = threadIdx.x & 31, w = threadIdx.x >> 5;
  int m0 = blockIdx.x * 64 + (w >> 2) * 32;
  int n0 = blockIdx.y * 128 + (w & 3) * 32;
  v8f acc[2][2] = {};
  for (int k0 = 0; k0 < K; k0 += 32) {
    v16bf a0 = load_frag_a(X, K, m0, k0, lane);
    v16bf a1 = load_frag_a(X, K, m0 + 16, k0, lane);
    v16bf b0 = load_frag_b_nt(W, K, n0, k0, lane);
    v16bf b1 = load_frag_b_nt(W, K, n0 + 16, k0, lane);
    acc[0][0] = wmma_bf16(a0, b0, acc[0][0]);
    acc[0][1] = wmma_bf16(a0, b1, acc[0][1]);
    acc[1][0] = wmma_bf16(a1, b0, acc[1][0]);
    acc[1][1] = wmma_bf16(a1, b1, acc[1][1]);
  }
  int hh = lane >> 4, nn = lane & 15;
  int b = m0 >> 10, nloc = m0 & (N_ - 1);
#pragma unroll
  for (int i = 0; i < 2; ++i)
#pragma unroll
    for (int j = 0; j < 2; ++j) {
      int row0 = nloc + 16 * i + 8 * hh;       // along N (contiguous in Yt)
      int col  = n0 + 16 * j + nn;             // along C
      alignas(16) unsigned short vb[8];
#pragma unroll
      for (int t = 0; t < 8; ++t) vb[t] = f2bf(acc[i][j][t]);
      *(uint4*)(Yt + ((size_t)b * Nt + col) * N_ + row0) = *(const uint4*)vb;
    }
}

// Transpose k heads: Yqk cols [512..1023] -> kT[b*H+h][d][n]  (bf16, ld=N_).
// Stages 32x64 chunks in LDS via async load-to-LDS (ASYNCcnt path).
__global__ __launch_bounds__(256) void k_ktrans(const unsigned short* __restrict__ Yqk,
                                                unsigned short* __restrict__ KT) {
  __shared__ __align__(16) unsigned short tile[32][HD_];
  int bh = blockIdx.y, b = bh / H_, h = bh % H_;
  const int ldq = H_ * HD_ * 2;  // 1024
  const unsigned short* kbase = Yqk + (size_t)b * N_ * ldq + H_ * HD_ + h * HD_;
  unsigned short* kt = KT + (size_t)bh * HD_ * N_;
  int t = threadIdx.x;
  for (int c = 0; c < 4; ++c) {
    int n0 = blockIdx.x * 128 + c * 32;
    {   // async-stage 32 rows x 64 bf16 (16B per thread)
      int row = t >> 3, seg = t & 7;
      const unsigned short* g = kbase + (size_t)(n0 + row) * ldq + seg * 8;
      unsigned lds_off = (unsigned)(uintptr_t)&tile[row][seg * 8];
      unsigned long long ga = (unsigned long long)(uintptr_t)g;
      asm volatile("global_load_async_to_lds_b128 %0, %1, off"
                   :: "v"(lds_off), "v"(ga) : "memory");
    }
    asm volatile("s_wait_asynccnt 0x0" ::: "memory");
    __syncthreads();
    {   // transposed write-out: contiguous 16B per thread
      int d = t >> 2, nseg = t & 3;
      alignas(16) unsigned short vb[8];
#pragma unroll
      for (int j = 0; j < 8; ++j) vb[j] = tile[nseg * 8 + j][d];
      *(uint4*)(kt + (size_t)d * N_ + n0 + nseg * 8) = *(const uint4*)vb;
    }
    __syncthreads();
  }
}

// Fused attention: per (b,h), 32 q-rows per WG.
//   S = scale * q @ k^T (f32 LDS) -> row softmax -> P (bf16 LDS) -> P @ k (NT via kT)
__global__ __launch_bounds__(256) void k_attn(const unsigned short* __restrict__ Yqk,
                                              const unsigned short* __restrict__ KT,
                                              float* __restrict__ AQK) {
  __shared__ __align__(16) float          S[32][N_];  // 128 KB
  __shared__ __align__(16) unsigned short P[32][N_];  //  64 KB
  const int ldq = H_ * HD_ * 2;  // 1024
  int bh = blockIdx.y, b = bh / H_, h = bh % H_;
  const unsigned short* qbase = Yqk + (size_t)b * N_ * ldq + h * HD_;
  const unsigned short* kbase = qbase + H_ * HD_;
  int m0 = blockIdx.x * 32;
  int lane = threadIdx.x & 31, w = threadIdx.x >> 5;
  const float scale = 0.125f;  // 64^-0.5

  // phase 1: S = scale * q @ k^T ; wave w covers S cols [w*128, w*128+128)
  {
    v16bf aq[2][2];
#pragma unroll
    for (int i = 0; i < 2; ++i)
#pragma unroll
      for (int ks = 0; ks < 2; ++ks)
        aq[i][ks] = load_frag_a(qbase, ldq, m0 + 16 * i, 32 * ks, lane);
    int hh = lane >> 4, nn = lane & 15;
    for (int nt = 0; nt < 8; ++nt) {
      int n0 = w * 128 + nt * 16;
      v8f s0 = {}, s1 = {};
#pragma unroll
      for (int ks = 0; ks < 2; ++ks) {
        v16bf bk = load_frag_b_nt(kbase, ldq, n0, 32 * ks, lane);
        s0 = wmma_bf16(aq[0][ks], bk, s0);
        s1 = wmma_bf16(aq[1][ks], bk, s1);
      }
#pragma unroll
      for (int t = 0; t < 8; ++t) {
        S[8 * hh + t][n0 + nn]      = s0[t] * scale;
        S[16 + 8 * hh + t][n0 + nn] = s1[t] * scale;
      }
    }
  }
  __syncthreads();

  // phase 2: row softmax (8 lanes per row, contiguous lanes in a wave)
  {
    int row = threadIdx.x >> 3, sub = threadIdx.x & 7;
    float mx = -3.0e38f;
    for (int j = sub * 128; j < sub * 128 + 128; ++j) mx = fmaxf(mx, S[row][j]);
#pragma unroll
    for (int o = 1; o < 8; o <<= 1) mx = fmaxf(mx, __shfl_xor(mx, o, 32));
    float sum = 0.0f;
    for (int j = sub * 128; j < sub * 128 + 128; ++j) {
      float e = __expf(S[row][j] - mx);
      S[row][j] = e;
      sum += e;
    }
#pragma unroll
    for (int o = 1; o < 8; o <<= 1) sum += __shfl_xor(sum, o, 32);
    float inv = 1.0f / sum;
    for (int j = sub * 128; j < sub * 128 + 128; ++j)
      P[row][j] = f2bf(S[row][j] * inv);
  }
  __syncthreads();

  // phase 3: AQK = P @ k, NT form with kT (contiguous B loads)
  {
    int mi = w >> 2, dj = w & 3;
    const unsigned short* ktb = KT + (size_t)bh * HD_ * N_;
    v8f acc = {};
    for (int k0 = 0; k0 < N_; k0 += 32) {
      v16bf a  = load_frag_a(&P[0][0], N_, 16 * mi, k0, lane);
      v16bf bk = load_frag_b_nt(ktb, N_, 16 * dj, k0, lane);
      acc = wmma_bf16(a, bk, acc);
    }
    float* out = AQK + ((size_t)b * N_ + m0 + 16 * mi) * C_ + h * HD_ + 16 * dj;
    int hh = lane >> 4, nn = lane & 15;
#pragma unroll
    for (int t = 0; t < 8; ++t) out[(size_t)(8 * hh + t) * C_ + nn] = acc[t];
  }
}

// av = aqk * softmax(aqk * lip_v over hd). One wave per (b,n,h) group of 64.
__global__ __launch_bounds__(256) void k_av(const float* __restrict__ AQK,
                                            const float* __restrict__ LIPV,
                                            unsigned short* __restrict__ AV,
                                            int total) {
  int gw = (blockIdx.x * blockDim.x + threadIdx.x) >> 5;
  if (gw >= total) return;
  int lane = threadIdx.x & 31;
  size_t base = (size_t)gw * HD_;
  float a0 = AQK[base + lane],  a1 = AQK[base + lane + 32];
  float l0 = LIPV[base + lane], l1 = LIPV[base + lane + 32];
  float t0 = a0 * l0, t1 = a1 * l1;
  float mx = fmaxf(t0, t1);
#pragma unroll
  for (int o = 1; o < 32; o <<= 1) mx = fmaxf(mx, __shfl_xor(mx, o, 32));
  float e0 = __expf(t0 - mx), e1 = __expf(t1 - mx);
  float s = e0 + e1;
#pragma unroll
  for (int o = 1; o < 32; o <<= 1) s += __shfl_xor(s, o, 32);
  float inv = 1.0f / s;
  AV[base + lane]      = f2bf(a0 * e0 * inv);
  AV[base + lane + 32] = f2bf(a1 * e1 * inv);
}

// av_attn[b] = sigmoid(2/temp * qs^T qs): NT gemm on qsT (contiguous loads).
__global__ __launch_bounds__(256) void k_se(const unsigned short* __restrict__ QST,
                                            unsigned short* __restrict__ AATT) {
  int b = blockIdx.z;
  const unsigned short* qst = QST + (size_t)b * C_ * N_;
  unsigned short* outb = AATT + (size_t)b * C_ * C_;
  int lane = threadIdx.x & 31, w = threadIdx.x >> 5;
  int c0 = blockIdx.x * 64, d0 = blockIdx.y * 64;
  int mi = w >> 1, pr = w & 1;
  v8f acc[2] = {};
  for (int k0 = 0; k0 < N_; k0 += 32) {
    v16bf a = load_frag_a(qst, N_, c0 + 16 * mi, k0, lane);
#pragma unroll
    for (int j = 0; j < 2; ++j) {
      v16bf bb = load_frag_b_nt(qst, N_, d0 + 16 * (pr * 2 + j), k0, lane);
      acc[j] = wmma_bf16(a, bb, acc[j]);
    }
  }
  const float c2t = 2.0f / 22.627416997969522f;  // 2 / sqrt(512)
  int hh = lane >> 4, nn = lane & 15;
#pragma unroll
  for (int j = 0; j < 2; ++j) {
    int row0 = c0 + 16 * mi, col0 = d0 + 16 * (pr * 2 + j);
#pragma unroll
    for (int t = 0; t < 8; ++t) {
      float x = acc[j][t] * c2t;
      outb[(size_t)(row0 + 8 * hh + t) * C_ + col0 + nn] =
          f2bf(1.0f / (1.0f + __expf(-x)));
    }
  }
}

// out[b] = av_out[b] @ av_attn[b]; av_attn is SYMMETRIC -> NT gemm, f32 out.
__global__ __launch_bounds__(256) void k_final(const unsigned short* __restrict__ AVOUT,
                                               const unsigned short* __restrict__ AATT,
                                               float* __restrict__ OUT) {
  int b = blockIdx.z;
  const unsigned short* A  = AVOUT + (size_t)b * N_ * C_;
  const unsigned short* Bm = AATT + (size_t)b * C_ * C_;
  float* O = OUT + (size_t)b * N_ * C_;
  int lane = threadIdx.x & 31, w = threadIdx.x >> 5;
  int m0 = blockIdx.x * 64, n0 = blockIdx.y * 64;
  int mi = w >> 1, pr = w & 1;
  v8f acc[2] = {};
  for (int k0 = 0; k0 < C_; k0 += 32) {
    v16bf a = load_frag_a(A, C_, m0 + 16 * mi, k0, lane);
#pragma unroll
    for (int j = 0; j < 2; ++j) {
      v16bf bb = load_frag_b_nt(Bm, C_, n0 + 16 * (pr * 2 + j), k0, lane);
      acc[j] = wmma_bf16(a, bb, acc[j]);
    }
  }
  int hh = lane >> 4, nn = lane & 15;
#pragma unroll
  for (int j = 0; j < 2; ++j) {
    int row0 = m0 + 16 * mi, col0 = n0 + 16 * (pr * 2 + j);
#pragma unroll
    for (int t = 0; t < 8; ++t)
      O[(size_t)(row0 + 8 * hh + t) * C_ + col0 + nn] = acc[j][t];
  }
}

// ----------------------------- host launcher -------------------------------
extern "C" void kernel_launch(void* const* d_in, const int* in_sizes, int n_in,
                              void* d_out, int out_size, void* d_ws, size_t ws_size,
                              hipStream_t stream) {
  const float* audia  = (const float*)d_in[0];
  const float* lip    = (const float*)d_in[1];
  const float* W_qkv  = (const float*)d_in[2];
  const float* W_lip  = (const float*)d_in[3];
  const float* W_proj = (const float*)d_in[4];
  const float* b_proj = (const float*)d_in[5];
  const float* W_se   = (const float*)d_in[6];
  float* out = (float*)d_out;

  unsigned char* ws = (unsigned char*)d_ws;
  size_t off = 0;
  auto take = [&](size_t bytes) -> void* {
    void* p = (void*)(ws + off);
    off += (bytes + 255) & ~(size_t)255;
    return p;
  };
  unsigned short* a_bf     = (unsigned short*)take((size_t)M_ * C_ * 2);
  unsigned short* l_bf     = (unsigned short*)take((size_t)M_ * C_ * 2);
  unsigned short* wqk_bf   = (unsigned short*)take((size_t)1024 * C_ * 2);
  unsigned short* wlip_bf  = (unsigned short*)take((size_t)C_ * C_ * 2);
  unsigned short* wproj_bf = (unsigned short*)take((size_t)C_ * C_ * 2);
  unsigned short* wse_bf   = (unsigned short*)take((size_t)C_ * C_ * 2);
  unsigned short* yqk_bf   = (unsigned short*)take((size_t)M_ * 1024 * 2);
  unsigned short* kt_bf    = (unsigned short*)take((size_t)B_ * H_ * HD_ * N_ * 2);
  float*          lipv_f   = (float*)take((size_t)M_ * C_ * 4);
  float*          aqk_f    = (float*)take((size_t)M_ * C_ * 4);
  unsigned short* av_bf    = (unsigned short*)take((size_t)M_ * C_ * 2);
  unsigned short* avout_bf = (unsigned short*)take((size_t)M_ * C_ * 2);
  unsigned short* qst_bf   = (unsigned short*)take((size_t)B_ * C_ * N_ * 2);
  unsigned short* aatt_bf  = (unsigned short*)take((size_t)B_ * C_ * C_ * 2);
  (void)ws_size; (void)in_sizes; (void)n_in; (void)out_size;

  auto cvt = [&](const float* src, unsigned short* dst, int n) {
    k_f2bf<<<(n + 255) / 256, 256, 0, stream>>>(src, dst, n);
  };
  cvt(audia, a_bf, M_ * C_);
  cvt(lip, l_bf, M_ * C_);
  cvt(W_qkv, wqk_bf, 1024 * C_);     // q,k rows only; v is dead
  cvt(W_lip, wlip_bf, C_ * C_);
  cvt(W_proj, wproj_bf, C_ * C_);
  cvt(W_se, wse_bf, C_ * C_);

  // qk projection: [8192,1024] = audia @ Wqk^T
  k_gemm_nt<true, false><<<dim3(M_ / 64, 1024 / 128), 256, 0, stream>>>(
      a_bf, wqk_bf, yqk_bf, nullptr, M_, 1024, C_);
  // kT[b*H+h][d][n] for NT attention phase 3 (async-to-LDS staged transpose)
  k_ktrans<<<dim3(N_ / 128, B_ * H_), 256, 0, stream>>>(yqk_bf, kt_bf);
  // lip_v: [8192,512] f32
  k_gemm_nt<false, false><<<dim3(M_ / 64, C_ / 128), 256, 0, stream>>>(
      l_bf, wlip_bf, lipv_f, nullptr, M_, C_, C_);
  // fused attention -> aqk f32 [B,N,C] ((h,d) packed into C)
  k_attn<<<dim3(N_ / 32, B_ * H_), 256, 0, stream>>>(yqk_bf, kt_bf, aqk_f);
  // av = aqk * softmax(aqk * lip_v over hd)
  k_av<<<(M_ * H_) / 8, 256, 0, stream>>>(aqk_f, lipv_f, av_bf, M_ * H_);
  // av_out = av @ W_proj^T + b_proj
  k_gemm_nt<true, true><<<dim3(M_ / 64, C_ / 128), 256, 0, stream>>>(
      av_bf, wproj_bf, avout_bf, b_proj, M_, C_, C_);
  // qsT[b][c][n] = (audia @ W_se^T)^T  (transposed-output gemm)
  k_gemm_nt_tout<<<dim3(M_ / 64, C_ / 128), 256, 0, stream>>>(
      a_bf, wse_bf, qst_bf, C_, C_);
  // av_attn = sigmoid(2 * qs^T qs / sqrt(C))
  k_se<<<dim3(C_ / 64, C_ / 64, B_), 256, 0, stream>>>(qst_bf, aatt_bf);
  // out = av_out @ av_attn  (av_attn symmetric -> NT)
  k_final<<<dim3(N_ / 64, C_ / 64, B_), 256, 0, stream>>>(avout_bf, aatt_bf, out);
}
